// LSTMModel_80582176407802
// MI455X (gfx1250) — compile-verified, matrix operand order checked by reading, not run on hardware
//
#include <hip/hip_runtime.h>
#include <hip/hip_bf16.h>

typedef __attribute__((ext_vector_type(16))) __bf16 v16bf;
typedef __attribute__((ext_vector_type(8)))  float  v8f;
typedef __attribute__((ext_vector_type(4)))  float  fvec4;

#define F_IN   256
#define NH     128
#define TSTEPS 7
#define NG     512           // 4*NH
#define MROWS  32            // rows per block (2 WMMA M-tiles per wave)
#define XROW   (F_IN + 8)    // 264 elems -> 132 dwords, stride % 64 banks = 4
#define HROW   (NH + 8)      // 136 elems -> 68 dwords,  stride % 64 banks = 4

__device__ __forceinline__ unsigned short f2bf(float x) {
    unsigned int u = __float_as_uint(x);
    u += 0x7FFFu + ((u >> 16) & 1u);          // round-to-nearest-even
    return (unsigned short)(u >> 16);
}
__device__ __forceinline__ float fast_sig(float x) {
    return __builtin_amdgcn_rcpf(1.0f + __builtin_amdgcn_exp2f(-1.44269504f * x));
}
__device__ __forceinline__ float fast_tanh(float x) {
    // tanh(x) = 1 - 2/(exp(2x)+1)
    return 1.0f - 2.0f * __builtin_amdgcn_rcpf(1.0f + __builtin_amdgcn_exp2f(2.88539008f * x));
}
// Build a v16bf WMMA operand from two contiguous 16B chunks (ISA K-layout:
// per-lane VGPRs 0-3 = K[8*half .. +7], VGPRs 4-7 = K[16+8*half .. +7]).
__device__ __forceinline__ v16bf ldmat(const unsigned short* p) {
    union { uint4 q[2]; v16bf v; } u;
    u.q[0] = *reinterpret_cast<const uint4*>(p);
    u.q[1] = *reinterpret_cast<const uint4*>(p + 16);
    return u.v;
}
__device__ __forceinline__ v8f wmma_bf16(v16bf a, v16bf b, v8f c) {
    return __builtin_amdgcn_wmma_f32_16x16x32_bf16(false, a, false, b,
                                                   (short)0, c, false, false);
}

// ---------------- one-time weight transpose + f32->bf16 convert ----------------
__global__ void prep_weights(const float* __restrict__ W, const float* __restrict__ U,
                             const float* __restrict__ W2,
                             unsigned short* __restrict__ WT,
                             unsigned short* __restrict__ UT,
                             unsigned short* __restrict__ W2T) {
    int idx = blockIdx.x * 256 + threadIdx.x;
    if (idx < NG * F_IN) {                       // WT[g][f] = W[f][g]
        int g = idx / F_IN, f = idx - g * F_IN;
        WT[idx] = f2bf(W[f * NG + g]);
        return;
    }
    idx -= NG * F_IN;
    if (idx < NG * NH) {                         // UT[g][h] = U[h][g]
        int g = idx / NH, h = idx - g * NH;
        UT[idx] = f2bf(U[h * NG + g]);
        return;
    }
    idx -= NG * NH;
    if (idx < 64 * NH) {                         // W2T[o][h] = W2[h][o]
        int o = idx / NH, h = idx - o * NH;
        W2T[idx] = f2bf(W2[h * 64 + o]);
    }
}

// ---------------- fused LSTM (7 steps) + Dense(64, relu) ----------------
__global__ __launch_bounds__(128) void lstm_fused(
    const float* __restrict__ x, const float* __restrict__ b,
    const float* __restrict__ b2,
    const unsigned short* __restrict__ WT, const unsigned short* __restrict__ UT,
    const unsigned short* __restrict__ W2T, float* __restrict__ out)
{
    __shared__ unsigned short xs[TSTEPS * MROWS * XROW];  // 118,272 B
    __shared__ unsigned short hs[MROWS * HROW];           //   8,704 B

    const int tid  = threadIdx.x;
    const int wg   = tid >> 5;        // wave 0..3: owns h-cols [32*wg, 32*wg+32)
    const int lane = tid & 31;
    const int ln   = lane & 15;
    const int half = lane >> 4;
    const int r0   = blockIdx.x * MROWS;

    // ---- stage x tile: [32 rows][7 t][256 f] f32 -> bf16 LDS [t][m][f] ----
    {
        const fvec4* xg = reinterpret_cast<const fvec4*>(x + (size_t)r0 * (TSTEPS * F_IN));
        for (int it = 0; it < (MROWS * TSTEPS * F_IN) / (4 * 128); ++it) {  // 112
            int i4   = tid + it * 128;
            fvec4 v  = __builtin_nontemporal_load(&xg[i4]);   // streamed once
            int flat = i4 * 4;
            int m    = flat / (TSTEPS * F_IN);
            int rem  = flat - m * (TSTEPS * F_IN);
            int t    = rem >> 8;
            int f    = rem & 255;
            unsigned int lo = (unsigned int)f2bf(v[0]) | ((unsigned int)f2bf(v[1]) << 16);
            unsigned int hi = (unsigned int)f2bf(v[2]) | ((unsigned int)f2bf(v[3]) << 16);
            uint2 pk; pk.x = lo; pk.y = hi;
            *reinterpret_cast<uint2*>(&xs[t * (MROWS * XROW) + m * XROW + f]) = pk;
        }
        for (int i = tid; i < (MROWS * HROW) / 2; i += 128)   // h0 = 0
            reinterpret_cast<unsigned int*>(hs)[i] = 0u;
    }
    __syncthreads();

    // ---- per-wave tile pointers + biases (tiles: gate g, half-tile h2) ----
    const unsigned short* wrow[4][2];
    const unsigned short* urow[4][2];
    float bz[4][2];
    #pragma unroll
    for (int g = 0; g < 4; ++g)
        #pragma unroll
        for (int h2 = 0; h2 < 2; ++h2) {
            int n = g * NH + wg * 32 + h2 * 16 + ln;  // per-lane output column
            wrow[g][h2] = WT + (size_t)n * F_IN;
            urow[g][h2] = UT + (size_t)n * NH;
            bz[g][h2]   = b[n];
        }

    float cst[2][2][8];                                     // [row-tile][h2][elem]
    #pragma unroll
    for (int r = 0; r < 2; ++r)
        #pragma unroll
        for (int h2 = 0; h2 < 2; ++h2)
            #pragma unroll
            for (int e = 0; e < 8; ++e) cst[r][h2][e] = 0.0f;

    // ---- recurrence ----
    #pragma unroll 1
    for (int t = 0; t < TSTEPS; ++t) {
        v8f acc[2][4][2];                                   // [row-tile][gate][h2]
        #pragma unroll
        for (int r = 0; r < 2; ++r)
            #pragma unroll
            for (int g = 0; g < 4; ++g)
                #pragma unroll
                for (int h2 = 0; h2 < 2; ++h2)
                    #pragma unroll
                    for (int e = 0; e < 8; ++e) acc[r][g][h2][e] = bz[g][h2];

        // z += x_t @ W   (K = 256 -> 8 k-steps, 2 row-tiles share each B tile)
        const unsigned short* xa0 = xs + t * (MROWS * XROW) + ln * XROW + 8 * half;
        const unsigned short* xa1 = xa0 + 16 * XROW;
        #pragma unroll
        for (int kk = 0; kk < 8; ++kk) {
            v16bf a0 = ldmat(xa0 + kk * 32);
            v16bf a1 = ldmat(xa1 + kk * 32);
            #pragma unroll
            for (int g = 0; g < 4; ++g)
                #pragma unroll
                for (int h2 = 0; h2 < 2; ++h2) {
                    v16bf bm = ldmat(wrow[g][h2] + kk * 32 + 8 * half);
                    acc[0][g][h2] = wmma_bf16(a0, bm, acc[0][g][h2]);
                    acc[1][g][h2] = wmma_bf16(a1, bm, acc[1][g][h2]);
                }
        }
        // z += h_{t-1} @ U   (K = 128 -> 4 k-steps); h_0 == 0 so skip at t==0
        if (t > 0) {
            const unsigned short* ha0 = hs + ln * HROW + 8 * half;
            const unsigned short* ha1 = ha0 + 16 * HROW;
            #pragma unroll
            for (int kk = 0; kk < 4; ++kk) {
                v16bf a0 = ldmat(ha0 + kk * 32);
                v16bf a1 = ldmat(ha1 + kk * 32);
                #pragma unroll
                for (int g = 0; g < 4; ++g)
                    #pragma unroll
                    for (int h2 = 0; h2 < 2; ++h2) {
                        v16bf bm = ldmat(urow[g][h2] + kk * 32 + 8 * half);
                        acc[0][g][h2] = wmma_bf16(a0, bm, acc[0][g][h2]);
                        acc[1][g][h2] = wmma_bf16(a1, bm, acc[1][g][h2]);
                    }
            }
        }
        __syncthreads();   // everyone finished reading h_{t-1}

        // gates (wave-local: all 4 gate tiles share the C/D layout) + write h_t
        #pragma unroll
        for (int r = 0; r < 2; ++r)
            #pragma unroll
            for (int h2 = 0; h2 < 2; ++h2)
                #pragma unroll
                for (int e = 0; e < 8; ++e) {
                    float i_ = fast_sig(acc[r][0][h2][e]);
                    float f_ = fast_sig(acc[r][1][h2][e]);
                    float g_ = fast_tanh(acc[r][2][h2][e]);
                    float o_ = fast_sig(acc[r][3][h2][e]);
                    float c  = f_ * cst[r][h2][e] + i_ * g_;
                    cst[r][h2][e] = c;
                    float h  = o_ * fast_tanh(c);
                    hs[(r * 16 + e + 8 * half) * HROW + wg * 32 + h2 * 16 + ln] = f2bf(h);
                }
        __syncthreads();   // h_t visible to all waves
    }

    // ---- out = relu(h_T @ W2 + b2): each wave owns 16 of 64 columns ----
    {
        v8f acc2[2];
        float bb = b2[wg * 16 + ln];
        #pragma unroll
        for (int r = 0; r < 2; ++r)
            #pragma unroll
            for (int e = 0; e < 8; ++e) acc2[r][e] = bb;
        const unsigned short* ha0   = hs + ln * HROW + 8 * half;
        const unsigned short* ha1   = ha0 + 16 * HROW;
        const unsigned short* w2row = W2T + (size_t)(wg * 16 + ln) * NH + 8 * half;
        #pragma unroll
        for (int kk = 0; kk < 4; ++kk) {
            v16bf a0 = ldmat(ha0 + kk * 32);
            v16bf a1 = ldmat(ha1 + kk * 32);
            v16bf bm = ldmat(w2row + kk * 32);
            acc2[0] = wmma_bf16(a0, bm, acc2[0]);
            acc2[1] = wmma_bf16(a1, bm, acc2[1]);
        }
        #pragma unroll
        for (int r = 0; r < 2; ++r)
            #pragma unroll
            for (int e = 0; e < 8; ++e) {
                float v = acc2[r][e];
                v = v > 0.0f ? v : 0.0f;
                out[(size_t)(r0 + r * 16 + e + 8 * half) * 64 + wg * 16 + ln] = v;
            }
    }
}

extern "C" void kernel_launch(void* const* d_in, const int* in_sizes, int n_in,
                              void* d_out, int out_size, void* d_ws, size_t ws_size,
                              hipStream_t stream) {
    const float* x  = (const float*)d_in[0];
    const float* W  = (const float*)d_in[1];
    const float* U  = (const float*)d_in[2];
    const float* b  = (const float*)d_in[3];
    const float* W2 = (const float*)d_in[4];
    const float* b2 = (const float*)d_in[5];

    unsigned short* WT  = (unsigned short*)d_ws;            // 512*256 bf16
    unsigned short* UT  = WT + NG * F_IN;                   // 512*128 bf16
    unsigned short* W2T = UT + NG * NH;                     // 64*128  bf16

    const int prep_elems = NG * F_IN + NG * NH + 64 * NH;   // 204800
    prep_weights<<<(prep_elems + 255) / 256, 256, 0, stream>>>(W, U, W2, WT, UT, W2T);

    lstm_fused<<<16384 / MROWS, 128, 0, stream>>>(x, b, b2, WT, UT, W2T, (float*)d_out);
}
